// MultiHeadedAttention_19129784336725
// MI455X (gfx1250) — matrix operand order repaired and checked by reading, not compiled, not run on hardware
//
#include <hip/hip_runtime.h>

typedef __attribute__((ext_vector_type(16))) __bf16 v16bf;
typedef __attribute__((ext_vector_type(4)))  __bf16 v4bf;
typedef __attribute__((ext_vector_type(8)))  float  v8f;

#define H_  16
#define S_  2048
#define D_  1024
#define DH_ 64

// A-matrix (16-bit, 16x32) K index of packed pair j (even element), per lane-half hs.
__device__ __forceinline__ int kbaseA(int j, int hs) {
  return ((j >= 4) ? 16 : 0) + hs * 8 + (j & 3) * 2;
}
// B-matrix (16-bit, 32x16) K index of packed pair j, same K order as kbaseA.
__device__ __forceinline__ int kbaseB(int j, int hs) {
  return ((j >= 4) ? 16 : 0) + hs * 8 + (j & 3) * 2;
}

// MODE 0: A = bf16 [M,K], Out = f32 [M,N] flat            (final projection)
// MODE 1: A = f32  [M,K], Out = bf16 head-split [B,H,S,dh] (Q, K projections)
// MODE 2: A = f32  [M,K], Out = bf16 transposed [B,H,dh,S] (V projection)
// Block tile: 64(M) x 128(N), K-step 32. 128 threads = 4 waves, each wave 16M x 128N.
template <int MODE>
__global__ __launch_bounds__(128)
void gemm_wmma(const void* __restrict__ Ap, const float* __restrict__ W,
               const float* __restrict__ bias, void* __restrict__ Outp,
               int M, int N, int K) {
  __shared__ __bf16 lsA[64 * 32];
  __shared__ __bf16 lsB[128 * 32];
  const int tid  = threadIdx.x;
  const int wave = tid >> 5, lane = tid & 31;
  const int l16  = lane & 15, hs = lane >> 4;
  const int n0 = blockIdx.x * 128, m0 = blockIdx.y * 64;

  const float*  Af = (const float*)Ap;
  const __bf16* Ab = (const __bf16*)Ap;

  v8f acc[8];
#pragma unroll
  for (int t = 0; t < 8; ++t)
#pragma unroll
    for (int r = 0; r < 8; ++r) acc[t][r] = 0.0f;

  for (int kk = 0; kk < K; kk += 32) {
    // ---- stage A: 64x32 = 512 float4-groups, 4 per thread ----
#pragma unroll
    for (int i = 0; i < 4; ++i) {
      int g = i * 128 + tid;
      int r = g >> 3, c4 = (g & 7) * 4;
      if constexpr (MODE == 0) {
        *(uint2*)&lsA[r * 32 + c4] =
            *(const uint2*)&Ab[(size_t)(m0 + r) * K + kk + c4];
      } else {
        float4 f = *(const float4*)&Af[(size_t)(m0 + r) * K + kk + c4];
        v4bf b;
        b.x = (__bf16)f.x; b.y = (__bf16)f.y; b.z = (__bf16)f.z; b.w = (__bf16)f.w;
        *(v4bf*)&lsA[r * 32 + c4] = b;
      }
    }
    // ---- stage B (weights, always f32): 128x32 = 1024 groups, 8 per thread ----
#pragma unroll
    for (int i = 0; i < 8; ++i) {
      int g = i * 128 + tid;
      int r = g >> 3, c4 = (g & 7) * 4;
      float4 f = *(const float4*)&W[(size_t)(n0 + r) * K + kk + c4];
      v4bf b;
      b.x = (__bf16)f.x; b.y = (__bf16)f.y; b.z = (__bf16)f.z; b.w = (__bf16)f.w;
      *(v4bf*)&lsB[r * 32 + c4] = b;
    }
    __syncthreads();

    union { v16bf v; unsigned u[8]; } fa;
    const __bf16* arow = &lsA[(wave * 16 + l16) * 32];
#pragma unroll
    for (int j = 0; j < 8; ++j)
      fa.u[j] = *(const unsigned*)&arow[kbaseA(j, hs)];

#pragma unroll
    for (int t = 0; t < 8; ++t) {
      union { v16bf v; unsigned u[8]; } fb;
      const __bf16* brow = &lsB[(t * 16 + l16) * 32];
#pragma unroll
      for (int j = 0; j < 8; ++j)
        fb.u[j] = *(const unsigned*)&brow[kbaseB(j, hs)];
      acc[t] = __builtin_amdgcn_wmma_f32_16x16x32_bf16(
          false, fa.v, false, fb.v, (short)0, acc[t], false, false);
    }
    __syncthreads();
  }

  // ---- epilogue: bias + layout-specific store ----
  float*  Of = (float*)Outp;
  __bf16* Ob = (__bf16*)Outp;
#pragma unroll
  for (int t = 0; t < 8; ++t) {
    int col = n0 + t * 16 + l16;
    float bvv = bias[col];
#pragma unroll
    for (int r = 0; r < 8; ++r) {
      int row = m0 + wave * 16 + r + 8 * hs;
      float val = acc[t][r] + bvv;
      if constexpr (MODE == 0) {
        Of[(size_t)row * N + col] = val;
      } else {
        int b = row >> 11, s = row & 2047;
        int hh = col >> 6, dd = col & 63;
        size_t oidx;
        if constexpr (MODE == 1)
          oidx = ((size_t)(b * H_ + hh) * S_ + s) * DH_ + dd;
        else
          oidx = ((size_t)(b * H_ + hh) * DH_ + dd) * S_ + s;
        Ob[oidx] = (__bf16)val;
      }
    }
  }
}

// Flash attention. Q,K: [B,H,S,dh] bf16. V transposed: [B,H,dh,S] bf16.
// One wave per 16-query tile, key blocks of 32, f32 online softmax.
// Softmax row stats computed by per-row owner lanes after an LDS transpose
// (2 shuffles per block instead of a 64-op butterfly cascade).
__global__ __launch_bounds__(128)
void attn_wmma(const __bf16* __restrict__ Q, const __bf16* __restrict__ Kt,
               const __bf16* __restrict__ Vt, __bf16* __restrict__ Oa) {
  __shared__ float  lsS[4][16 * 36]; // padded stride 36 (bank-conflict free)
  __shared__ __bf16 lsP[4][16 * 32];
  const int tid  = threadIdx.x;
  const int wave = tid >> 5, lane = tid & 31;
  const int l16  = lane & 15, hs = lane >> 4;
  const int bh   = blockIdx.y;
  const int q0   = blockIdx.x * 64 + wave * 16;
  const size_t base = (size_t)bh * S_ * DH_;

  union Frag { v16bf v; unsigned u[8]; };

  Frag aq0, aq1; // Q tile, dh halves 0..31 / 32..63
  {
    const __bf16* qrow = &Q[base + (size_t)(q0 + l16) * DH_];
#pragma unroll
    for (int j = 0; j < 8; ++j) {
      aq0.u[j] = *(const unsigned*)&qrow[kbaseA(j, hs)];
      aq1.u[j] = *(const unsigned*)&qrow[32 + kbaseA(j, hs)];
    }
  }

  // Owner-lane softmax state: lane owns row (lane&15); both halves replicate.
  float mo = -__builtin_inff(), lo = 0.0f;
  v8f o[4];
#pragma unroll
  for (int t = 0; t < 4; ++t)
#pragma unroll
    for (int r = 0; r < 8; ++r) o[t][r] = 0.0f;

  const float scale = 0.125f; // 1/sqrt(64)

  for (int j0 = 0; j0 < S_; j0 += 32) {
    // ---- load K fragments (32 keys x 64 dh) ----
    Frag kb0[2], kb1[2];
#pragma unroll
    for (int t = 0; t < 2; ++t) {
      const __bf16* krow = &Kt[base + (size_t)(j0 + t * 16 + l16) * DH_];
#pragma unroll
      for (int jj = 0; jj < 8; ++jj) {
        kb0[t].u[jj] = *(const unsigned*)&krow[kbaseB(jj, hs)];
        kb1[t].u[jj] = *(const unsigned*)&krow[32 + kbaseB(jj, hs)];
      }
    }
    // ---- scores: 16q x 32k ----
    v8f sc[2];
#pragma unroll
    for (int t = 0; t < 2; ++t) {
      v8f c;
#pragma unroll
      for (int r = 0; r < 8; ++r) c[r] = 0.0f;
      c = __builtin_amdgcn_wmma_f32_16x16x32_bf16(false, aq0.v, false, kb0[t].v,
                                                  (short)0, c, false, false);
      c = __builtin_amdgcn_wmma_f32_16x16x32_bf16(false, aq1.v, false, kb1[t].v,
                                                  (short)0, c, false, false);
      sc[t] = c;
    }

    // ---- prefetch V fragments now; latency overlaps the softmax VALU ----
    Frag bv[4];
#pragma unroll
    for (int t = 0; t < 4; ++t) {
      const __bf16* vcol = &Vt[base + (size_t)(t * 16 + l16) * S_ + j0];
#pragma unroll
      for (int jj = 0; jj < 8; ++jj)
        bv[t].u[jj] = *(const unsigned*)&vcol[kbaseB(jj, hs)];
    }

    // ---- transpose scaled scores into LDS (C-layout -> row-major) ----
#pragma unroll
    for (int r = 0; r < 8; ++r) {
      int row = r + 8 * hs;
      lsS[wave][row * 36 + l16]      = sc[0][r] * scale;
      lsS[wave][row * 36 + 16 + l16] = sc[1][r] * scale;
    }
    asm volatile("s_wait_dscnt 0" ::: "memory");

    // ---- owner-lane softmax: lane handles row (l16), column half (hs) ----
    const float4* srow = (const float4*)&lsS[wave][l16 * 36 + hs * 16];
    float v[16];
#pragma unroll
    for (int i = 0; i < 4; ++i) {
      float4 f = srow[i];
      v[4 * i + 0] = f.x; v[4 * i + 1] = f.y; v[4 * i + 2] = f.z; v[4 * i + 3] = f.w;
    }
    float m16 = v[0];
#pragma unroll
    for (int i = 1; i < 16; ++i) m16 = fmaxf(m16, v[i]);
    m16 = fmaxf(m16, __shfl_xor(m16, 16, 32));   // combine column halves
    float mnew = fmaxf(mo, m16);
    float fsc  = __expf(mo - mnew);
    float s16  = 0.0f;
#pragma unroll
    for (int i = 0; i < 16; ++i) { v[i] = __expf(v[i] - mnew); s16 += v[i]; }
    s16 += __shfl_xor(s16, 16, 32);
    lo = lo * fsc + s16;
    mo = mnew;

    // ---- write P (bf16, packed pairs) back to LDS in row-major ----
    unsigned* pdst = (unsigned*)&lsP[wave][l16 * 32 + hs * 16];
#pragma unroll
    for (int i = 0; i < 8; ++i) {
      union { __bf16 b[2]; unsigned u; } w;
      w.b[0] = (__bf16)v[2 * i];
      w.b[1] = (__bf16)v[2 * i + 1];
      pdst[i] = w.u;
    }

    // ---- broadcast rescale factor to C-layout lanes, rescale O ----
    float fb[8];
#pragma unroll
    for (int r = 0; r < 8; ++r) fb[r] = __shfl(fsc, r + 8 * hs, 32);
#pragma unroll
    for (int t = 0; t < 4; ++t)
#pragma unroll
      for (int r = 0; r < 8; ++r) o[t][r] *= fb[r];

    asm volatile("s_wait_dscnt 0" ::: "memory");

    // ---- O += P @ V ----
    Frag ap;
    {
      const __bf16* prow = &lsP[wave][l16 * 32];
#pragma unroll
      for (int jj = 0; jj < 8; ++jj)
        ap.u[jj] = *(const unsigned*)&prow[kbaseA(jj, hs)];
    }
#pragma unroll
    for (int t = 0; t < 4; ++t)
      o[t] = __builtin_amdgcn_wmma_f32_16x16x32_bf16(
          false, ap.v, false, bv[t].v, (short)0, o[t], false, false);
  }

  // ---- normalize + store bf16 [B, S, H*dh] ----
  const int b = bh >> 4, hh = bh & 15;
  float linv_own = 1.0f / lo;
#pragma unroll
  for (int r = 0; r < 8; ++r) {
    float inv = __shfl(linv_own, r + 8 * hs, 32);
    int s = q0 + r + 8 * hs;
#pragma unroll
    for (int t = 0; t < 4; ++t) {
      int dd = t * 16 + l16;
      Oa[((size_t)(b * S_ + s) * D_) + hh * DH_ + dd] = (__bf16)(o[t][r] * inv);
    }
  }
}

extern "C" void kernel_launch(void* const* d_in, const int* in_sizes, int n_in,
                              void* d_out, int out_size, void* d_ws, size_t ws_size,
                              hipStream_t stream) {
  (void)in_sizes; (void)n_in; (void)out_size; (void)ws_size;
  const float* query = (const float*)d_in[0];
  const float* key   = (const float*)d_in[1];
  const float* value = (const float*)d_in[2];
  const float* Wq = (const float*)d_in[3];
  const float* bq = (const float*)d_in[4];
  const float* Wk = (const float*)d_in[5];
  const float* bk = (const float*)d_in[6];
  const float* Wv = (const float*)d_in[7];
  const float* bvp = (const float*)d_in[8];
  const float* Wo = (const float*)d_in[9];
  const float* bo = (const float*)d_in[10];
  float* out = (float*)d_out;

  const int B = 2;
  const int M = B * S_, N = D_, K = D_;
  const size_t nQ = (size_t)B * H_ * S_ * DH_; // 4,194,304 elems (8 MB bf16)

  __bf16* Qb = (__bf16*)d_ws;
  __bf16* Kb = Qb + nQ;
  __bf16* Vt = Kb + nQ;
  __bf16* Ab = Vt + nQ; // attention output, bf16 [B,S,D]

  dim3 gg(N / 128, M / 64);
  gemm_wmma<1><<<gg, 128, 0, stream>>>(query, Wq, bq, Qb, M, N, K);
  gemm_wmma<1><<<gg, 128, 0, stream>>>(key,   Wk, bk, Kb, M, N, K);
  gemm_wmma<2><<<gg, 128, 0, stream>>>(value, Wv, bvp, Vt, M, N, K);
  attn_wmma<<<dim3(S_ / 64, B * H_), 128, 0, stream>>>(Qb, Kb, Vt, Ab);
  gemm_wmma<0><<<gg, 128, 0, stream>>>(Ab, Wo, bo, out, M, N, K);
}
